// DK_ECA_14559939133869
// MI455X (gfx1250) — matrix operand order repaired and checked by reading, not compiled, not run on hardware
//
#include <hip/hip_runtime.h>
#include <math.h>

typedef __attribute__((ext_vector_type(16))) _Float16 v16h;
typedef __attribute__((ext_vector_type(8)))  float    v8f;
typedef __attribute__((ext_vector_type(4)))  float    v4f;

#define B_  16
#define C_  512
#define L_  8192
#define H_  64
#define BASE_K 5
#define CENTER 4

// ---------------------------------------------------------------------------
// Kernel 1: gap[row] = mean_l x[row][l], row = b*C_ + c  (8192 rows x 8192 f32)
// One 256-thread block per row; each thread reads 8 float4 (b128 loads).
// Regular-temporal loads: deliberately populate L2 so kernel 3 can re-hit the
// tail of x (192 MB L2 vs 256 MB tensor).
// ---------------------------------------------------------------------------
__global__ __launch_bounds__(256) void gap_kernel(const float* __restrict__ x,
                                                  float* __restrict__ gap) {
    const int row = blockIdx.x;
    const float4* __restrict__ x4 =
        reinterpret_cast<const float4*>(x) + (size_t)row * (L_ / 4);
    float sum = 0.f;
#pragma unroll
    for (int i = 0; i < 8; ++i) {
        float4 v = x4[i * 256 + threadIdx.x];
        sum += (v.x + v.y) + (v.z + v.w);
    }
#pragma unroll
    for (int off = 16; off > 0; off >>= 1)
        sum += __shfl_down(sum, off, 32);
    __shared__ float red[8];
    const int wid  = threadIdx.x >> 5;
    const int lane = threadIdx.x & 31;
    if (lane == 0) red[wid] = sum;
    __syncthreads();
    if (threadIdx.x == 0) {
        float t = 0.f;
#pragma unroll
        for (int i = 0; i < 8; ++i) t += red[i];
        gap[row] = t * (1.0f / (float)L_);
    }
}

// ---------------------------------------------------------------------------
// Kernel 2: tiny MLP via WMMA (one workgroup), then per-(b,c) windowed
// average of gap + sigmoid -> s[16*512].
//   H(16x64) = GAP(16x512) @ W1^T(512x64): M=16, K=512, N=64
//   -> 4 N-tiles of 16, 16 K-steps of 32, v_wmma_f32_16x16x32_f16.
// ---------------------------------------------------------------------------
__global__ __launch_bounds__(256) void mlp_scale_kernel(
    const float* __restrict__ gap, const float* __restrict__ w1,
    const float* __restrict__ b1,  const float* __restrict__ w2,
    const float* __restrict__ b2,  float* __restrict__ s) {

    __shared__ float h_lds[B_][H_];   // gelu(GAP@W1^T + b1)
    __shared__ int   k_arr[B_];       // per-batch dynamic kernel size

    if (threadIdx.x < 32) {           // one full wave32, EXEC all-ones
        const int lane = threadIdx.x;
        const int m    = lane & 15;   // A-matrix row (batch)
        const int hi   = lane >> 4;   // half-wave selector

#pragma unroll
        for (int nt = 0; nt < 4; ++nt) {
            v8f acc = {};
            const int ncol = nt * 16 + (lane & 15);   // B/C/D column (hidden)
            for (int kb = 0; kb < 16; ++kb) {
                const int kbase = kb * 32;
                v16h a, bm;
                // A fragment: 16-bit A 16x32 layout (lane m, hi selects K+8 / K+24)
#pragma unroll
                for (int i = 0; i < 8; ++i) {
                    a[i]     = (_Float16)gap[m * C_ + kbase + hi * 8 + i];
                    a[8 + i] = (_Float16)gap[m * C_ + kbase + 16 + hi * 8 + i];
                }
                // B fragment: same K order as A (hi selects K+8 / K+24)
#pragma unroll
                for (int e = 0; e < 16; ++e)
                    bm[e] = (_Float16)w1[ncol * C_ + kbase + (e < 8 ? hi * 8 + e : 16 + hi * 8 + (e - 8))];
                acc = __builtin_amdgcn_wmma_f32_16x16x32_f16(
                    /*neg_a=*/false, a, /*neg_b=*/false, bm,
                    /*c_mod=*/(short)0, acc, /*reuse_a=*/false, /*reuse_b=*/false);
            }
            // C/D layout: acc[r] = D[M = 8*hi + r][N = ncol]
#pragma unroll
            for (int r = 0; r < 8; ++r) {
                float v = acc[r] + b1[ncol];
                float g = 0.5f * v * (1.0f + erff(v * 0.70710678118f)); // exact gelu
                h_lds[8 * hi + r][ncol] = g;
            }
        }
    }
    __syncthreads();

    if (threadIdx.x < B_) {
        const int b = threadIdx.x;
        float d = b2[0];
#pragma unroll
        for (int n = 0; n < H_; ++n) d += h_lds[b][n] * w2[n];
        float delta = tanhf(d) * 2.0f;
        float dk = rintf(delta);                    // round-to-nearest-even
        dk = fminf(3.0f, fmaxf(-3.0f, dk));
        k_arr[b] = (BASE_K + (int)dk) | 1;          // odd k in {3,5,7,9}
    }
    __syncthreads();

    for (int idx = threadIdx.x; idx < B_ * C_; idx += 256) {
        const int b = idx >> 9;      // / C_
        const int c = idx & (C_ - 1);
        const int k = k_arr[b];
        const int half = (k - 1) >> 1;
        float y = 0.f;
        for (int o = -half; o <= half; ++o) {
            const int j = c + o;
            if (j >= 0 && j < C_) y += gap[b * C_ + j];  // zero-padded window
        }
        y *= (1.0f / (float)k);
        s[idx] = 1.0f / (1.0f + expf(-y));
    }
}

// ---------------------------------------------------------------------------
// Kernel 3: out[b][c][l] = x[b][c][l] * s[b*C_+c].  b128 in/out, 4/thread.
// Blocks traverse x in REVERSE so the tail of x (still resident in the
// 192 MB L2 after kernel 1's ascending pass) is re-read before eviction.
// Output stores are non-temporal (write-once stream, don't evict x lines);
// x loads are non-temporal too (dead after this pass, don't re-allocate).
// ---------------------------------------------------------------------------
__global__ __launch_bounds__(256) void scale_kernel(const float* __restrict__ x,
                                                    const float* __restrict__ s,
                                                    float* __restrict__ out,
                                                    int num_blocks) {
    const v4f* __restrict__ x4 = reinterpret_cast<const v4f*>(x);
    v4f* __restrict__ o4 = reinterpret_cast<v4f*>(out);
    const int rb = num_blocks - 1 - (int)blockIdx.x;       // reversed traversal
    const size_t base = (size_t)rb * 1024 + threadIdx.x;
#pragma unroll
    for (int t = 0; t < 4; ++t) {
        const size_t i = base + (size_t)t * 256;           // < 16,777,216 float4
        const int row = (int)(i >> 11);                    // 2048 float4 per row
        const float sv = s[row];
        v4f v = __builtin_nontemporal_load(&x4[i]);
        v *= sv;
        __builtin_nontemporal_store(v, &o4[i]);
    }
}

// ---------------------------------------------------------------------------
extern "C" void kernel_launch(void* const* d_in, const int* in_sizes, int n_in,
                              void* d_out, int out_size, void* d_ws, size_t ws_size,
                              hipStream_t stream) {
    const float* x  = (const float*)d_in[0];   // (16, 512, 8192)
    const float* w1 = (const float*)d_in[1];   // (64, 512)
    const float* b1 = (const float*)d_in[2];   // (64,)
    const float* w2 = (const float*)d_in[3];   // (1, 64)
    const float* b2 = (const float*)d_in[4];   // (1,)
    float* out = (float*)d_out;

    float* gap = (float*)d_ws;                 // 8192 floats
    float* sc  = gap + B_ * C_;                // 8192 floats

    const int nblk3 = (B_ * C_ * L_) / 4 / 1024;   // 16384

    gap_kernel<<<B_ * C_, 256, 0, stream>>>(x, gap);
    mlp_scale_kernel<<<1, 256, 0, stream>>>(gap, w1, b1, w2, b2, sc);
    scale_kernel<<<nblk3, 256, 0, stream>>>(x, sc, out, nblk3);
}
